// CrossAttention_3693671875330
// MI455X (gfx1250) — matrix operand order repaired and checked by reading, not compiled, not run on hardware
//
#include <hip/hip_runtime.h>
#include <hip/hip_bf16.h>

typedef __bf16 bf16;
typedef __attribute__((ext_vector_type(4)))  bf16  bf16x4;
typedef __attribute__((ext_vector_type(8)))  bf16  bf16x8;
typedef __attribute__((ext_vector_type(16))) bf16  v16bf;
typedef __attribute__((ext_vector_type(8)))  float v8f;

#define NB  4
#define LQ  2048
#define LKV 2048
#define QD  1024
#define KVD 768
#define H   16
#define HD  64
#define OD  1024

// ---- WMMA operand loaders (contiguous k order k = 16*(L/16) + s for A and B, wave32) ----

// A-matrix 16x32 bf16: lane L holds row L%16; half=L/16 selects K-run
// {half*16..+15} (matching load_b16): two 16B chunks.
__device__ __forceinline__ v16bf load_a16(const bf16* base, int lda, int lane) {
  const bf16* p = base + (size_t)(lane & 15) * lda + ((lane >> 4) << 4);
  bf16x8 c0 = *(const bf16x8*)p;
  bf16x8 c1 = *(const bf16x8*)(p + 8);
  return __builtin_shufflevector(c0, c1, 0,1,2,3,4,5,6,7,8,9,10,11,12,13,14,15);
}

// B-matrix 32x16 bf16 from a transposed (N x K row-major) source:
// lane L holds column N=L%16, K-run (L/16)*16 .. +15: one contiguous 32B load.
__device__ __forceinline__ v16bf load_b16(const bf16* baseT, int ldb, int lane) {
  const bf16* p = baseT + (size_t)(lane & 15) * ldb + ((lane >> 4) << 4);
  return *(const v16bf*)p;
}

__device__ __forceinline__ v8f wmma_bf16(v16bf a, v16bf b, v8f c) {
  return __builtin_amdgcn_wmma_f32_16x16x32_bf16(false, a, false, b,
                                                 (short)0, c, false, false);
}

// Row reductions across the 16-lane group holding one C-tile row.
__device__ __forceinline__ float rmax16(float v) {
  v = fmaxf(v, __shfl_xor(v, 1));
  v = fmaxf(v, __shfl_xor(v, 2));
  v = fmaxf(v, __shfl_xor(v, 4));
  v = fmaxf(v, __shfl_xor(v, 8));
  return v;
}
__device__ __forceinline__ float rsum16(float v) {
  v += __shfl_xor(v, 1);
  v += __shfl_xor(v, 2);
  v += __shfl_xor(v, 4);
  v += __shfl_xor(v, 8);
  return v;
}

// ---- vectorized fp32 -> bf16 convert (n divisible by 4) ----
__global__ void k_cvt(const float* __restrict__ src, bf16* __restrict__ dst, int n4) {
  int i = blockIdx.x * blockDim.x + threadIdx.x;
  if (i < n4) {
    float4 f = ((const float4*)src)[i];
    bf16x4 o = { (bf16)f.x, (bf16)f.y, (bf16)f.z, (bf16)f.w };
    ((bf16x4*)dst)[i] = o;
  }
}

// W (in x out) f32 -> Wt (out x in) bf16
__global__ void k_tr(const float* __restrict__ W, bf16* __restrict__ Wt,
                     int in_dim, int out_dim) {
  int i = blockIdx.x * blockDim.x + threadIdx.x;
  if (i < in_dim * out_dim) {
    int o = i / in_dim, k = i - o * in_dim;
    Wt[i] = (bf16)W[(size_t)k * out_dim + o];
  }
}

// ---- projection GEMM: X(8192 x kdim) @ Wt -> per-head layouts ----
// 2x4 register blocking: one wave computes a 32x64 output block.
// vmode==0: dst[(nb*H+h)*L + l][d]   (Q with scale, K)
// vmode==1: dst[(nb*H+h)*HD + d][l]  (V transposed for PV B-operand)
__global__ void k_proj(const bf16* __restrict__ A, const bf16* __restrict__ Wt,
                       bf16* __restrict__ dst, int kdim, float scale, int vmode) {
  int lane = threadIdx.x & 31;
  int wv = blockIdx.x * 8 + (threadIdx.x >> 5);   // 4096 waves
  int m0 = (wv >> 4) << 5;                        // 256 M-blocks of 32 rows
  int n0 = (wv & 15) << 6;                        // 16  N-blocks of 64 cols
  const bf16* A0 = A  + (size_t)m0 * kdim;
  const bf16* B0 = Wt + (size_t)n0 * kdim;
  v8f acc[2][4] = {};
  int kiter = kdim >> 5;
  for (int kb = 0; kb < kiter; ++kb) {
    int ko = kb * 32;
    __builtin_prefetch(A0 + ko + 32, 0, 1);
    v16bf a0 = load_a16(A0 + ko,                 kdim, lane);
    v16bf a1 = load_a16(A0 + 16 * kdim + ko,     kdim, lane);
    v16bf b0 = load_b16(B0 + ko,                 kdim, lane);
    v16bf b1 = load_b16(B0 + 16 * kdim + ko,     kdim, lane);
    v16bf b2 = load_b16(B0 + 32 * kdim + ko,     kdim, lane);
    v16bf b3 = load_b16(B0 + 48 * kdim + ko,     kdim, lane);
    acc[0][0] = wmma_bf16(a0, b0, acc[0][0]);
    acc[0][1] = wmma_bf16(a0, b1, acc[0][1]);
    acc[0][2] = wmma_bf16(a0, b2, acc[0][2]);
    acc[0][3] = wmma_bf16(a0, b3, acc[0][3]);
    acc[1][0] = wmma_bf16(a1, b0, acc[1][0]);
    acc[1][1] = wmma_bf16(a1, b1, acc[1][1]);
    acc[1][2] = wmma_bf16(a1, b2, acc[1][2]);
    acc[1][3] = wmma_bf16(a1, b3, acc[1][3]);
  }
  int half = lane >> 4;
#pragma unroll
  for (int mi = 0; mi < 2; ++mi) {
#pragma unroll
    for (int ni = 0; ni < 4; ++ni) {
      int col = n0 + ni * 16 + (lane & 15);
      int h = col >> 6, d = col & 63;
#pragma unroll
      for (int r = 0; r < 8; ++r) {
        int row = m0 + mi * 16 + r + (half << 3);
        int nb = row >> 11, l = row & 2047;
        float v = acc[mi][ni][r] * scale;
        size_t idx = vmode ? ((size_t)((nb * H + h) * HD + d) * LKV + l)
                           : ((size_t)((nb * H + h) * LKV + l) * HD + d);
        dst[idx] = (bf16)v;
      }
    }
  }
}

// ---- flash attention: one wave = 32 q rows x full Lkv ----
// K/V B-operands loaded once per 32-kv chunk, reused across both q row-blocks.
__global__ void k_attn(const bf16* __restrict__ Qs, const bf16* __restrict__ Ks,
                       const bf16* __restrict__ Vt, bf16* __restrict__ Ob) {
  __shared__ bf16 pb[8][2][16 * 32];                // per-wave P bounce (2 KB each)
  int lane = threadIdx.x & 31;
  int widx = threadIdx.x >> 5;
  int w    = blockIdx.x * 8 + widx;                 // 4096 waves total
  int bh   = w >> 6;                                // 64 batch*head
  int qb2  = w & 63;                                // 64 blocks of 32 q rows
  int half = lane >> 4;

  const bf16* Qh = Qs + ((size_t)bh * LQ + qb2 * 32) * HD;
  const bf16* Kh = Ks + (size_t)bh * LKV * HD;
  const bf16* Vh = Vt + (size_t)bh * HD * LKV;

  v16bf aQ[2][2];
  aQ[0][0] = load_a16(Qh,                HD, lane);
  aQ[0][1] = load_a16(Qh + 32,           HD, lane);
  aQ[1][0] = load_a16(Qh + 16 * HD,      HD, lane);
  aQ[1][1] = load_a16(Qh + 16 * HD + 32, HD, lane);

  float m[2][8], l[2][8];
  v8f o[2][4] = {};
#pragma unroll
  for (int qi = 0; qi < 2; ++qi)
#pragma unroll
    for (int r = 0; r < 8; ++r) { m[qi][r] = -3.0e38f; l[qi][r] = 0.0f; }

  for (int j = 0; j < LKV / 32; ++j) {
    const bf16* Kb = Kh + (size_t)(j * 32) * HD;
    v16bf b00 = load_b16(Kb,                HD, lane);
    v16bf b01 = load_b16(Kb + 32,           HD, lane);
    v16bf b10 = load_b16(Kb + 16 * HD,      HD, lane);
    v16bf b11 = load_b16(Kb + 16 * HD + 32, HD, lane);

    v8f s[2][2] = {};
#pragma unroll
    for (int qi = 0; qi < 2; ++qi) {
      s[qi][0] = wmma_bf16(aQ[qi][0], b00, s[qi][0]);
      s[qi][0] = wmma_bf16(aQ[qi][1], b01, s[qi][0]);
      s[qi][1] = wmma_bf16(aQ[qi][0], b10, s[qi][1]);
      s[qi][1] = wmma_bf16(aQ[qi][1], b11, s[qi][1]);
    }

    // online softmax per q row-block
#pragma unroll
    for (int qi = 0; qi < 2; ++qi) {
      bf16* P = pb[widx][qi];
#pragma unroll
      for (int r = 0; r < 8; ++r) {
        float sm   = rmax16(fmaxf(s[qi][0][r], s[qi][1][r]));
        float nm   = fmaxf(m[qi][r], sm);
        float corr = __expf(m[qi][r] - nm);
        float p0   = __expf(s[qi][0][r] - nm);
        float p1   = __expf(s[qi][1][r] - nm);
        l[qi][r] = l[qi][r] * corr + rsum16(p0 + p1);
        m[qi][r] = nm;
        o[qi][0][r] *= corr; o[qi][1][r] *= corr;
        o[qi][2][r] *= corr; o[qi][3][r] *= corr;
        int rowoff = (r + (half << 3)) * 32 + (lane & 15);
        P[rowoff]      = (bf16)p0;
        P[rowoff + 16] = (bf16)p1;
      }
    }

    // O += P @ V : V operands shared across both q blocks
    v16bf aP0 = load_a16(pb[widx][0], 32, lane);
    v16bf aP1 = load_a16(pb[widx][1], 32, lane);
    const bf16* Vb = Vh + j * 32;
    v16bf bv0 = load_b16(Vb,            LKV, lane);
    v16bf bv1 = load_b16(Vb + 16 * LKV, LKV, lane);
    v16bf bv2 = load_b16(Vb + 32 * LKV, LKV, lane);
    v16bf bv3 = load_b16(Vb + 48 * LKV, LKV, lane);
    o[0][0] = wmma_bf16(aP0, bv0, o[0][0]);
    o[0][1] = wmma_bf16(aP0, bv1, o[0][1]);
    o[0][2] = wmma_bf16(aP0, bv2, o[0][2]);
    o[0][3] = wmma_bf16(aP0, bv3, o[0][3]);
    o[1][0] = wmma_bf16(aP1, bv0, o[1][0]);
    o[1][1] = wmma_bf16(aP1, bv1, o[1][1]);
    o[1][2] = wmma_bf16(aP1, bv2, o[1][2]);
    o[1][3] = wmma_bf16(aP1, bv3, o[1][3]);
  }

  int nb = bh >> 4, h = bh & 15;
  int colb = h * HD + (lane & 15);
#pragma unroll
  for (int qi = 0; qi < 2; ++qi) {
#pragma unroll
    for (int r = 0; r < 8; ++r) {
      float inv = 1.0f / l[qi][r];
      int lrow = qb2 * 32 + qi * 16 + r + (half << 3);
      size_t base = ((size_t)(nb * LQ + lrow)) * OD + colb;
      Ob[base]      = (bf16)(o[qi][0][r] * inv);
      Ob[base + 16] = (bf16)(o[qi][1][r] * inv);
      Ob[base + 32] = (bf16)(o[qi][2][r] * inv);
      Ob[base + 48] = (bf16)(o[qi][3][r] * inv);
    }
  }
}

// ---- final projection + bias, fp32 out; 2x4 register blocking ----
__global__ void k_out(const bf16* __restrict__ A, const bf16* __restrict__ Wt,
                      const float* __restrict__ bias, float* __restrict__ out) {
  int lane = threadIdx.x & 31;
  int wv = blockIdx.x * 8 + (threadIdx.x >> 5);
  int m0 = (wv >> 4) << 5;
  int n0 = (wv & 15) << 6;
  const bf16* A0 = A  + (size_t)m0 * OD;
  const bf16* B0 = Wt + (size_t)n0 * OD;
  v8f acc[2][4] = {};
  for (int kb = 0; kb < OD / 32; ++kb) {
    int ko = kb * 32;
    __builtin_prefetch(A0 + ko + 32, 0, 1);
    v16bf a0 = load_a16(A0 + ko,             OD, lane);
    v16bf a1 = load_a16(A0 + 16 * OD + ko,   OD, lane);
    v16bf b0 = load_b16(B0 + ko,             OD, lane);
    v16bf b1 = load_b16(B0 + 16 * OD + ko,   OD, lane);
    v16bf b2 = load_b16(B0 + 32 * OD + ko,   OD, lane);
    v16bf b3 = load_b16(B0 + 48 * OD + ko,   OD, lane);
    acc[0][0] = wmma_bf16(a0, b0, acc[0][0]);
    acc[0][1] = wmma_bf16(a0, b1, acc[0][1]);
    acc[0][2] = wmma_bf16(a0, b2, acc[0][2]);
    acc[0][3] = wmma_bf16(a0, b3, acc[0][3]);
    acc[1][0] = wmma_bf16(a1, b0, acc[1][0]);
    acc[1][1] = wmma_bf16(a1, b1, acc[1][1]);
    acc[1][2] = wmma_bf16(a1, b2, acc[1][2]);
    acc[1][3] = wmma_bf16(a1, b3, acc[1][3]);
  }
  int half = lane >> 4;
#pragma unroll
  for (int mi = 0; mi < 2; ++mi) {
#pragma unroll
    for (int ni = 0; ni < 4; ++ni) {
      int col = n0 + ni * 16 + (lane & 15);
      float bs = bias[col];
#pragma unroll
      for (int r = 0; r < 8; ++r) {
        int row = m0 + mi * 16 + r + (half << 3);
        out[(size_t)row * OD + col] = acc[mi][ni][r] + bs;
      }
    }
  }
}

extern "C" void kernel_launch(void* const* d_in, const int* in_sizes, int n_in,
                              void* d_out, int out_size, void* d_ws, size_t ws_size,
                              hipStream_t stream) {
  (void)in_sizes; (void)n_in; (void)out_size; (void)ws_size;
  const float* query = (const float*)d_in[0];
  const float* kv    = (const float*)d_in[1];
  const float* Wq    = (const float*)d_in[2];
  const float* Wk    = (const float*)d_in[3];
  const float* Wv    = (const float*)d_in[4];
  const float* Wo    = (const float*)d_in[5];
  const float* bo    = (const float*)d_in[6];
  float* out = (float*)d_out;

  char* ws = (char*)d_ws;
  size_t off = 0;
  auto alloc = [&](size_t bytes) -> char* {
    char* p = ws + off;
    off += (bytes + 255) & ~(size_t)255;
    return p;
  };
  bf16* qb  = (bf16*)alloc((size_t)NB * LQ  * QD  * 2);
  bf16* kvb = (bf16*)alloc((size_t)NB * LKV * KVD * 2);
  bf16* wqT = (bf16*)alloc((size_t)QD * QD  * 2);
  bf16* wkT = (bf16*)alloc((size_t)QD * KVD * 2);
  bf16* wvT = (bf16*)alloc((size_t)QD * KVD * 2);
  bf16* woT = (bf16*)alloc((size_t)OD * QD  * 2);
  bf16* Qs  = (bf16*)alloc((size_t)NB * H * LQ  * HD * 2);
  bf16* Ksb = (bf16*)alloc((size_t)NB * H * LKV * HD * 2);
  bf16* Vtb = (bf16*)alloc((size_t)NB * H * HD * LKV * 2);
  bf16* Obf = (bf16*)alloc((size_t)NB * LQ * OD * 2);

  int nq4 = NB * LQ * QD / 4;
  k_cvt<<<(nq4 + 255) / 256, 256, 0, stream>>>(query, qb, nq4);
  int nk4 = NB * LKV * KVD / 4;
  k_cvt<<<(nk4 + 255) / 256, 256, 0, stream>>>(kv, kvb, nk4);
  k_tr<<<(QD * QD  + 255) / 256, 256, 0, stream>>>(Wq, wqT, QD,  QD);
  k_tr<<<(KVD * QD + 255) / 256, 256, 0, stream>>>(Wk, wkT, KVD, QD);
  k_tr<<<(KVD * QD + 255) / 256, 256, 0, stream>>>(Wv, wvT, KVD, QD);
  k_tr<<<(QD * OD  + 255) / 256, 256, 0, stream>>>(Wo, woT, QD,  OD);

  const float scale = 0.125f;  // hd^-0.5, hd=64
  k_proj<<<512, 256, 0, stream>>>(qb,  wqT, Qs,  QD,  scale, 0);
  k_proj<<<512, 256, 0, stream>>>(kvb, wkT, Ksb, KVD, 1.0f,  0);
  k_proj<<<512, 256, 0, stream>>>(kvb, wvT, Vtb, KVD, 1.0f,  1);

  k_attn<<<512, 256, 0, stream>>>(Qs, Ksb, Vtb, Obf);

  k_out<<<512, 256, 0, stream>>>(Obf, woT, bo, out);
}
